// SetAbstraction_90202903151206
// MI455X (gfx1250) — matrix operand order repaired and checked by reading, not compiled, not run on hardware
//
#include <hip/hip_runtime.h>
#include <hip/hip_bf16.h>

// ---------------- problem constants ----------------
constexpr int B_   = 8;
constexpr int N_   = 8192;
constexpr int CIN_ = 64;
constexpr int NP_  = 2048;   // N/STRIDE
constexpr int NS_  = 32;     // nsample
constexpr float R2_  = 0.1f * 0.1f;
constexpr float EPS_ = 1e-5f;
constexpr int K1PAD = 96;    // 3+64=67 padded to 3 K-tiles of 32
constexpr int C1_ = 64;
constexpr int C2_ = 128;

typedef __attribute__((ext_vector_type(16))) __bf16 bf16x16;
typedef __attribute__((ext_vector_type(8)))  __bf16 bf16x8;
typedef __attribute__((ext_vector_type(8)))  float  f32x8;

// load a 16-element bf16 WMMA fragment as two 16-byte halves
__device__ __forceinline__ bf16x16 ld_frag(const __bf16* lo, const __bf16* hi) {
    bf16x8 a = *(const bf16x8*)lo;
    bf16x8 b = *(const bf16x8*)hi;
    bf16x16 r;
#pragma unroll
    for (int e = 0; e < 8; ++e) { r[e] = a[e]; r[e + 8] = b[e]; }
    return r;
}

// ---------------- 1) furthest point sampling: one block per batch ----------------
__global__ __launch_bounds__(1024) void fps_kernel(const float* __restrict__ p,
                                                   int* __restrict__ fidx) {
    const int b = blockIdx.x;
    const int t = threadIdx.x;
    const int lane = t & 31, wid = t >> 5;
    __shared__ float rv[32];
    __shared__ int   ri[32];
    __shared__ float lX, lY, lZ;

    float px[8], py[8], pz[8], dist[8];
#pragma unroll
    for (int j = 0; j < 8; ++j) {
        int pid = t + 1024 * j;
        const float* pp = &p[(b * N_ + pid) * 3];
        px[j] = pp[0]; py[j] = pp[1]; pz[j] = pp[2];
        dist[j] = 1e10f;
    }
    if (t == 0) {
        fidx[b * NP_] = 0;
        const float* pp = &p[b * N_ * 3];
        lX = pp[0]; lY = pp[1]; lZ = pp[2];
    }
    __syncthreads();

    for (int it = 1; it < NP_; ++it) {
        float lx = lX, ly = lY, lz = lZ;
        float bv = -1.0f; int bi = 0;
#pragma unroll
        for (int j = 0; j < 8; ++j) {
            float dx = px[j] - lx, dy = py[j] - ly, dz = pz[j] - lz;
            float d = dx * dx + dy * dy + dz * dz;
            dist[j] = fminf(dist[j], d);
            int pid = t + 1024 * j;
            if (dist[j] > bv || (dist[j] == bv && pid < bi)) { bv = dist[j]; bi = pid; }
        }
        // wave32 argmax reduce (first-index tie-break like jnp.argmax)
#pragma unroll
        for (int mk = 16; mk >= 1; mk >>= 1) {
            float ov = __shfl_xor(bv, mk, 32);
            int   oi = __shfl_xor(bi, mk, 32);
            if (ov > bv || (ov == bv && oi < bi)) { bv = ov; bi = oi; }
        }
        if (lane == 0) { rv[wid] = bv; ri[wid] = bi; }
        __syncthreads();
        if (wid == 0) {
            float v = rv[lane]; int i = ri[lane];
#pragma unroll
            for (int mk = 16; mk >= 1; mk >>= 1) {
                float ov = __shfl_xor(v, mk, 32);
                int   oi = __shfl_xor(i, mk, 32);
                if (ov > v || (ov == v && oi < i)) { v = ov; i = oi; }
            }
            if (lane == 0) {
                fidx[b * NP_ + it] = i;
                const float* pp = &p[(b * N_ + i) * 3];
                lX = pp[0]; lY = pp[1]; lZ = pp[2];
            }
        }
        __syncthreads();
    }
}

// ---------------- 2) gather new_p (first output) ----------------
__global__ void gather_newp(const float* __restrict__ p, const int* __restrict__ fidx,
                            float* __restrict__ newp) {
    int t = blockIdx.x * 256 + threadIdx.x;
    if (t >= B_ * NP_) return;
    int b = t / NP_;
    int id = fidx[t];
    const float* pp = &p[(b * N_ + id) * 3];
    newp[t * 3 + 0] = pp[0];
    newp[t * 3 + 1] = pp[1];
    newp[t * 3 + 2] = pp[2];
}

// ---------------- 3) transpose f [B,64,N] -> ft [B,N,64] ----------------
__global__ void transpose_f(const float* __restrict__ f, float* __restrict__ ft) {
    __shared__ float tile[32][33];
    int b = blockIdx.z;
    int n0 = blockIdx.x * 32, c0 = blockIdx.y * 32;
    int tx = threadIdx.x, ty = threadIdx.y;      // 32 x 8
#pragma unroll
    for (int i = 0; i < 32; i += 8)
        tile[ty + i][tx] = f[(b * CIN_ + c0 + ty + i) * N_ + n0 + tx];
    __syncthreads();
#pragma unroll
    for (int i = 0; i < 32; i += 8)
        ft[(b * N_ + n0 + ty + i) * CIN_ + c0 + tx] = tile[tx][ty + i];
}

// ---------------- 4) ball query: first 32 indices within radius (ascending) ----------------
__global__ __launch_bounds__(256) void ball_query_kernel(const float* __restrict__ p,
                                                         const float* __restrict__ newp,
                                                         int* __restrict__ grp) {
    constexpr int CHUNK = 2048;
    __shared__ float pb[CHUNK * 3];     // 24 KB
    __shared__ int   gbuf[256 * 33];    // 33.8 KB, stride 33 avoids bank conflicts
    int b  = blockIdx.y;
    int mq = blockIdx.x * 256 + threadIdx.x;
    float qx = newp[(b * NP_ + mq) * 3 + 0];
    float qy = newp[(b * NP_ + mq) * 3 + 1];
    float qz = newp[(b * NP_ + mq) * 3 + 2];
    int cnt = 0;
    for (int c0 = 0; c0 < N_; c0 += CHUNK) {
        __syncthreads();
        for (int i = threadIdx.x; i < CHUNK * 3; i += 256)
            pb[i] = p[b * N_ * 3 + c0 * 3 + i];
        __syncthreads();
        for (int j = 0; j < CHUNK && cnt < NS_; ++j) {
            float dx = pb[3 * j + 0] - qx;
            float dy = pb[3 * j + 1] - qy;
            float dz = pb[3 * j + 2] - qz;
            if (dx * dx + dy * dy + dz * dz < R2_) {
                gbuf[threadIdx.x * 33 + cnt] = c0 + j;
                ++cnt;
            }
        }
    }
    int first = gbuf[threadIdx.x * 33];     // cnt >= 1 always (query point is in p)
#pragma unroll
    for (int k = 0; k < NS_; ++k)
        grp[(b * NP_ + mq) * NS_ + k] = (k < cnt) ? gbuf[threadIdx.x * 33 + k] : first;
}

// ---------------- 5) weights -> bf16, K-padded ----------------
__global__ void prep_weights(const float* __restrict__ w1, const float* __restrict__ w2,
                             __bf16* __restrict__ w1b, __bf16* __restrict__ w2b) {
    int t = blockIdx.x * 256 + threadIdx.x;
    if (t < C1_ * K1PAD) {
        int m = t / K1PAD, k = t % K1PAD;
        w1b[t] = (__bf16)((k < 67) ? w1[m * 67 + k] : 0.0f);
    }
    if (t < C2_ * C1_) w2b[t] = (__bf16)w2[t];
}

// ---------------- 6) group + MLP(67->64->128, bf16 WMMA, f32 acc) + max ----------------
constexpr int WAVES = 4;   // groups per block, one wave32 per group

__global__ __launch_bounds__(WAVES * 32) void mlp_kernel(
    const float* __restrict__ p, const float* __restrict__ newp,
    const int* __restrict__ grp, const float* __restrict__ ft,
    const __bf16* __restrict__ w1b, const __bf16* __restrict__ w2b,
    const float* __restrict__ g1, const float* __restrict__ b1,
    const float* __restrict__ m1, const float* __restrict__ v1,
    const float* __restrict__ g2, const float* __restrict__ b2,
    const float* __restrict__ m2, const float* __restrict__ v2,
    float* __restrict__ outF) {
    __shared__ __bf16 feat[WAVES][NS_][K1PAD];  // 24 KB: B-matrix layer 1 [sample][chan]
    __shared__ __bf16 x1s[WAVES][NS_][C1_];     // 16 KB: B-matrix layer 2

    const int w    = threadIdx.x >> 5;
    const int lane = threadIdx.x & 31;
    const int g    = blockIdx.x * WAVES + w;
    const int b    = g / NP_;
    const int mq   = g % NP_;
    const int half = lane >> 4;
    const int nl   = lane & 15;

    // ---- stage features: lane == sample index ----
    {
        int pid = grp[g * NS_ + lane];
        float nx = newp[(b * NP_ + mq) * 3 + 0];
        float ny = newp[(b * NP_ + mq) * 3 + 1];
        float nz = newp[(b * NP_ + mq) * 3 + 2];
        const float* pp = &p[(b * N_ + pid) * 3];
        feat[w][lane][0] = (__bf16)(pp[0] - nx);
        feat[w][lane][1] = (__bf16)(pp[1] - ny);
        feat[w][lane][2] = (__bf16)(pp[2] - nz);
        const float4* fp = (const float4*)&ft[(b * N_ + pid) * CIN_];
#pragma unroll
        for (int c = 0; c < CIN_ / 4; ++c) {
            float4 fv = fp[c];
            feat[w][lane][3 + 4 * c + 0] = (__bf16)fv.x;
            feat[w][lane][3 + 4 * c + 1] = (__bf16)fv.y;
            feat[w][lane][3 + 4 * c + 2] = (__bf16)fv.z;
            feat[w][lane][3 + 4 * c + 3] = (__bf16)fv.w;
        }
#pragma unroll
        for (int k = 67; k < K1PAD; ++k) feat[w][lane][k] = (__bf16)0.0f;
    }
    __syncthreads();

    // ---- layer 1: D[64x32] = W1[64x96] x feat^T[96x32] ----
    f32x8 zero8 = {0.f, 0.f, 0.f, 0.f, 0.f, 0.f, 0.f, 0.f};
    f32x8 acc[4][2];
#pragma unroll
    for (int mt = 0; mt < 4; ++mt)
#pragma unroll
        for (int nt = 0; nt < 2; ++nt) acc[mt][nt] = zero8;

#pragma unroll
    for (int kt = 0; kt < 3; ++kt) {
        bf16x16 bf[2];
#pragma unroll
        for (int nt = 0; nt < 2; ++nt) {
            const __bf16* base = &feat[w][16 * nt + nl][32 * kt + 8 * half];
            bf[nt] = ld_frag(base, base + 16);
        }
#pragma unroll
        for (int mt = 0; mt < 4; ++mt) {
            const __bf16* ab = &w1b[(16 * mt + nl) * K1PAD + 32 * kt + 8 * half];
            bf16x16 af = ld_frag(ab, ab + 16);
#pragma unroll
            for (int nt = 0; nt < 2; ++nt)
                acc[mt][nt] = __builtin_amdgcn_wmma_f32_16x16x32_bf16(
                    false, af, false, bf[nt], (short)0, acc[mt][nt], false, false);
        }
    }

    // ---- BN + ReLU, write layer-2 B matrix to LDS ----
#pragma unroll
    for (int mt = 0; mt < 4; ++mt)
#pragma unroll
        for (int r = 0; r < 8; ++r) {
            int ch = 16 * mt + 8 * half + r;
            float sc = g1[ch] * rsqrtf(v1[ch] + EPS_);
            float bs = b1[ch] - m1[ch] * sc;
#pragma unroll
            for (int nt = 0; nt < 2; ++nt) {
                float y = fmaxf(acc[mt][nt][r] * sc + bs, 0.0f);
                x1s[w][16 * nt + nl][ch] = (__bf16)y;
            }
        }
    __syncthreads();

    // ---- layer 2: D[128x32] = W2[128x64] x x1[64x32] ----
    f32x8 acc2[8][2];
#pragma unroll
    for (int mt = 0; mt < 8; ++mt)
#pragma unroll
        for (int nt = 0; nt < 2; ++nt) acc2[mt][nt] = zero8;

#pragma unroll
    for (int kt = 0; kt < 2; ++kt) {
        bf16x16 bf[2];
#pragma unroll
        for (int nt = 0; nt < 2; ++nt) {
            const __bf16* base = &x1s[w][16 * nt + nl][32 * kt + 8 * half];
            bf[nt] = ld_frag(base, base + 16);
        }
#pragma unroll
        for (int mt = 0; mt < 8; ++mt) {
            const __bf16* ab = &w2b[(16 * mt + nl) * C1_ + 32 * kt + 8 * half];
            bf16x16 af = ld_frag(ab, ab + 16);
#pragma unroll
            for (int nt = 0; nt < 2; ++nt)
                acc2[mt][nt] = __builtin_amdgcn_wmma_f32_16x16x32_bf16(
                    false, af, false, bf[nt], (short)0, acc2[mt][nt], false, false);
        }
    }

    // ---- BN + ReLU + max over 32 samples, write f_out[b][ch][mq] ----
#pragma unroll
    for (int mt = 0; mt < 8; ++mt)
#pragma unroll
        for (int r = 0; r < 8; ++r) {
            int ch = 16 * mt + 8 * half + r;
            float sc = g2[ch] * rsqrtf(v2[ch] + EPS_);
            float bs = b2[ch] - m2[ch] * sc;
            float y0 = fmaxf(acc2[mt][0][r] * sc + bs, 0.0f);
            float y1 = fmaxf(acc2[mt][1][r] * sc + bs, 0.0f);
            float vmax = fmaxf(y0, y1);
            // reduce over 16 lanes of this half-group (samples)
#pragma unroll
            for (int mk = 1; mk <= 8; mk <<= 1)
                vmax = fmaxf(vmax, __shfl_xor(vmax, mk, 32));
            if (nl == 0)
                outF[(b * C2_ + ch) * NP_ + mq] = vmax;
        }
}

// ---------------- host launcher ----------------
extern "C" void kernel_launch(void* const* d_in, const int* in_sizes, int n_in,
                              void* d_out, int out_size, void* d_ws, size_t ws_size,
                              hipStream_t stream) {
    const float* p  = (const float*)d_in[0];
    const float* f  = (const float*)d_in[1];
    const float* w1 = (const float*)d_in[2];
    const float* g1 = (const float*)d_in[3];
    const float* b1 = (const float*)d_in[4];
    const float* m1 = (const float*)d_in[5];
    const float* v1 = (const float*)d_in[6];
    const float* w2 = (const float*)d_in[7];
    const float* g2 = (const float*)d_in[8];
    const float* b2 = (const float*)d_in[9];
    const float* m2 = (const float*)d_in[10];
    const float* v2 = (const float*)d_in[11];

    float* out  = (float*)d_out;
    float* newp = out;                                   // [B, NP, 3]
    float* outF = out + (size_t)B_ * NP_ * 3;            // [B, 128, NP]

    char* ws = (char*)d_ws;
    int* fidx = (int*)ws;                 ws += (size_t)B_ * NP_ * sizeof(int);
    int* grp  = (int*)ws;                 ws += (size_t)B_ * NP_ * NS_ * sizeof(int);
    float* ft = (float*)ws;               ws += (size_t)B_ * N_ * CIN_ * sizeof(float);
    __bf16* w1b = (__bf16*)ws;            ws += (size_t)C1_ * K1PAD * sizeof(__bf16);
    __bf16* w2b = (__bf16*)ws;

    fps_kernel<<<B_, 1024, 0, stream>>>(p, fidx);
    gather_newp<<<(B_ * NP_ + 255) / 256, 256, 0, stream>>>(p, fidx, newp);
    transpose_f<<<dim3(N_ / 32, CIN_ / 32, B_), dim3(32, 8), 0, stream>>>(f, ft);
    ball_query_kernel<<<dim3(NP_ / 256, B_), 256, 0, stream>>>(p, newp, grp);
    prep_weights<<<(C2_ * C1_ + 255) / 256, 256, 0, stream>>>(w1, w2, w1b, w2b);
    mlp_kernel<<<(B_ * NP_) / WAVES, WAVES * 32, 0, stream>>>(
        p, newp, grp, ft, w1b, w2b, g1, b1, m1, v1, g2, b2, m2, v2, outF);
}
